// Guesser_2559800508528
// MI455X (gfx1250) — matrix operand order repaired and checked by reading, not compiled, not run on hardware
//
#include <hip/hip_runtime.h>
#include <hip/hip_bf16.h>

// ---------------------------------------------------------------------------
// Guesser fused pipeline for gfx1250 (MI455X), bf16 WMMA + f32 accumulate.
//
// B=4096, T=32, K=16, D=512, H=1024, H2=2048.
// Split the concat-matmuls:
//   h  = relu(g_hat@W1[:512] + X@W1[512:] + b1); e = h.w2 ; softmax over T
//   h2 = relu(x_hat@M1[:512] + G@M1[512:] + c1); logit = h2.m2 + c2
// One workgroup (256 thr = 8 waves) handles BB=4 batch elements end-to-end.
// ---------------------------------------------------------------------------

typedef __attribute__((ext_vector_type(16))) __bf16 v16bf;
typedef __attribute__((ext_vector_type(8)))  float  v8f;

#define D_   512
#define T_   32
#define KSP  16
#define H1   1024
#define H2   2048
#define BB   4
#define KT   16          // K-tiles: 512 / 32

__device__ __forceinline__ unsigned short f2bf(float f) {
  unsigned u = __float_as_uint(f);
  unsigned r = u + 0x7FFFu + ((u >> 16) & 1u);   // round-to-nearest-even
  return (unsigned short)(r >> 16);
}
__device__ __forceinline__ float bf2f(unsigned short h) {
  return __uint_as_float(((unsigned)h) << 16);
}

// A-matrix (16x32, 16-bit) per-lane element index inside a 512-entry tile.
// ISA 7.12.2: lanes 0-15 M=0-15 hold K pairs {0..7,16..23}, lanes 16-31 hold
// K pairs {8..15,24..31}.
__device__ __forceinline__ int frag_a(int m, int kl) {
  int p = kl & 1;
  int half, v;
  if      (kl < 8)  { half = 0; v = kl >> 1; }
  else if (kl < 16) { half = 1; v = (kl - 8) >> 1; }
  else if (kl < 24) { half = 0; v = 4 + ((kl - 16) >> 1); }
  else              { half = 1; v = 4 + ((kl - 24) >> 1); }
  return (half * 16 + m) * 16 + (v * 2 + p);
}

// ---------------------------------------------------------------------------
// Weight pack: W (row-major KxN f32, rows [k0,k0+512)) -> bf16 tiles in the
// WMMA B-matrix (32x16) lane layout: lane = h*16 + n, element j:
// k_local = 8h + j (j<8), 16 + 8h + (j-8) (j>=8).  Tile order: [nt][kt][lane][16].
// ---------------------------------------------------------------------------
__global__ void pack_w_kernel(const float* __restrict__ W,
                              unsigned short* __restrict__ out,
                              int nTiles, int k0, int N) {
  int gid = blockIdx.x * 256 + threadIdx.x;
  int total = nTiles * KT * 512;
  if (gid >= total) return;
  int e    = gid & 511;
  int kt   = (gid >> 9) & 15;
  int nt   = gid >> 13;
  int lane = e >> 4, j = e & 15;
  int n      = nt * 16 + (lane & 15);
  int k_loc  = (j < 8) ? (lane >> 4) * 8 + j : 16 + (lane >> 4) * 8 + (j - 8);
  int k      = k0 + kt * 32 + k_loc;
  out[gid] = f2bf(W[(size_t)k * N + n]);
}

// ---------------------------------------------------------------------------
// Main fused kernel: 1024 blocks x 256 threads (8 wave32).
// ---------------------------------------------------------------------------
__global__ __launch_bounds__(256) void guesser_fused_kernel(
    const float* __restrict__ X, const float* __restrict__ G,
    const float* __restrict__ b1, const float* __restrict__ w2,
    const float* __restrict__ c1, const float* __restrict__ m2,
    const float* __restrict__ c2p,
    const unsigned short* __restrict__ W1a, const unsigned short* __restrict__ W1b,
    const unsigned short* __restrict__ M1a, const unsigned short* __restrict__ M1b,
    float* __restrict__ out) {

  // ---- LDS (~297 KB of the 320 KB WGP budget) ----
  __shared__ unsigned short sX [8 * KT * 512];   // 128 KB: X rows, A-layout
  __shared__ unsigned short sG [4 * KT * 512];   //  64 KB: G rows, A-layout
  __shared__ unsigned short sGH[KT * 512];       //  16 KB: g_hat tile (rows 0..3)
  __shared__ unsigned short sXH[KT * 512];       //  16 KB: x_hat tile (rows 0..3)
  __shared__ float su[BB][H1];                   //  16 KB: W1_top^T g_hat
  __shared__ float sv[BB][H2];                   //  32 KB: M1_top^T x_hat
  __shared__ float sb1[H1], sw2[H1];             //   8 KB
  __shared__ float sc1[H2], sm2[H2];             //  16 KB
  __shared__ float se[BB][T_];                   // scores -> alpha (in place)
  __shared__ float slog[BB][KSP];                // logit accumulators

  const int tid  = threadIdx.x;
  const int lane = tid & 31;
  const int wv   = tid >> 5;
  const int half = lane >> 4;
  const int b0   = blockIdx.x * BB;

  // ---- stage 0: constants to LDS, zero accumulators / padded tiles ----
  for (int i = tid; i < H1; i += 256) { sb1[i] = b1[i]; sw2[i] = w2[i]; }
  for (int i = tid; i < H2; i += 256) { sc1[i] = c1[i]; sm2[i] = m2[i]; }
  for (int i = tid; i < KT * 512; i += 256) { sGH[i] = 0; sXH[i] = 0; }
  if (tid < BB * T_)  se[tid >> 5][tid & 31] = 0.f;
  if (tid < BB * KSP) slog[tid >> 4][tid & 15] = 0.f;

  // ---- stage 1: pack X and G (bf16, A layout); g_hat = mean_K(G) ----
  for (int o = tid; o < BB * T_ * D_; o += 256) {
    int row = o >> 9, d = o & 511;
    sX[(((row >> 4) * KT) + (d >> 5)) * 512 + frag_a(row & 15, d & 31)] =
        f2bf(X[(size_t)b0 * T_ * D_ + o]);
  }
  for (int o = tid; o < BB * KSP * D_; o += 256) {
    int row = o >> 9, d = o & 511;
    sG[(((row >> 4) * KT) + (d >> 5)) * 512 + frag_a(row & 15, d & 31)] =
        f2bf(G[(size_t)b0 * KSP * D_ + o]);
  }
  for (int o = tid; o < BB * D_; o += 256) {
    int bb = o >> 9, d = o & 511;
    const float* gp = G + (size_t)(b0 + bb) * KSP * D_ + d;
    float s = 0.f;
#pragma unroll
    for (int k = 0; k < KSP; ++k) s += gp[k * D_];
    sGH[(d >> 5) * 512 + frag_a(bb, d & 31)] = f2bf(s * (1.f / 16.f));
  }
  __syncthreads();

  // ---- stage 2: u = W1[:512]^T g_hat  (one padded 16-row tile) ----
  for (int nt = wv * 8; nt < wv * 8 + 8; ++nt) {
    v8f acc = {};
    for (int kt = 0; kt < KT; ++kt) {
      v16bf a = *(const v16bf*)&sGH[kt * 512 + lane * 16];
      v16bf b = *(const v16bf*)&W1a[((size_t)(nt * KT + kt)) * 512 + lane * 16];
      acc = __builtin_amdgcn_wmma_f32_16x16x32_bf16(false, a, false, b,
                                                    (short)0, acc, false, false);
    }
    if (lane < 16) {
      int n = nt * 16 + lane;
#pragma unroll
      for (int r = 0; r < BB; ++r) su[r][n] = acc[r];
    }
  }
  __syncthreads();

  // ---- stage 3: e[b,t] = w2 . relu(X@W1[512:] + u + b1) ----
  {
    float el[8][8];
#pragma unroll
    for (int rt = 0; rt < 8; ++rt)
#pragma unroll
      for (int r = 0; r < 8; ++r) el[rt][r] = 0.f;

    for (int ntb = 0; ntb < 8; ++ntb) {      // waves split N: B tile read once
      int nt = wv * 8 + ntb;
      v8f acc[8];
#pragma unroll
      for (int rt = 0; rt < 8; ++rt) acc[rt] = (v8f){};
      for (int kt = 0; kt < KT; ++kt) {
        v16bf b = *(const v16bf*)&W1b[((size_t)(nt * KT + kt)) * 512 + lane * 16];
#pragma unroll
        for (int rt = 0; rt < 8; ++rt) {
          v16bf a = *(const v16bf*)&sX[(rt * KT + kt) * 512 + lane * 16];
          acc[rt] = __builtin_amdgcn_wmma_f32_16x16x32_bf16(
              false, a, false, b, (short)0, acc[rt], false, false);
        }
      }
      int n = nt * 16 + (lane & 15);
      float w2n = sw2[n];
#pragma unroll
      for (int rt = 0; rt < 8; ++rt) {
        float uv = su[rt >> 1][n] + sb1[n];
#pragma unroll
        for (int r = 0; r < 8; ++r) {
          float h = fmaxf(acc[rt][r] + uv, 0.f);
          el[rt][r] += h * w2n;
        }
      }
    }
#pragma unroll
    for (int rt = 0; rt < 8; ++rt)
#pragma unroll
      for (int r = 0; r < 8; ++r) {
        int row = rt * 16 + (r + 8 * half);   // C layout: M = r + 8*(lane>=16)
        atomicAdd(&se[row >> 5][row & 31], el[rt][r]);
      }
  }
  __syncthreads();

  // ---- stage 4: softmax over T, then x_hat = alpha @ X ----
  if (tid < BB) {
    float mx = -1e30f;
    for (int t = 0; t < T_; ++t) mx = fmaxf(mx, se[tid][t]);
    float s = 0.f;
    for (int t = 0; t < T_; ++t) { float ev = __expf(se[tid][t] - mx); se[tid][t] = ev; s += ev; }
    float inv = 1.f / s;
    for (int t = 0; t < T_; ++t) se[tid][t] *= inv;
  }
  __syncthreads();
  for (int o = tid; o < BB * D_; o += 256) {
    int bb = o >> 9, d = o & 511;
    int kt = d >> 5, kl = d & 31;
    float s = 0.f;
    for (int t = 0; t < T_; ++t) {
      int row = bb * T_ + t;
      s += se[bb][t] * bf2f(sX[((row >> 4) * KT + kt) * 512 + frag_a(row & 15, kl)]);
    }
    sXH[kt * 512 + frag_a(bb, kl)] = f2bf(s);
  }
  __syncthreads();

  // ---- stage 5: v = M1[:512]^T x_hat ----
  for (int nt = wv * 16; nt < wv * 16 + 16; ++nt) {
    v8f acc = {};
    for (int kt = 0; kt < KT; ++kt) {
      v16bf a = *(const v16bf*)&sXH[kt * 512 + lane * 16];
      v16bf b = *(const v16bf*)&M1a[((size_t)(nt * KT + kt)) * 512 + lane * 16];
      acc = __builtin_amdgcn_wmma_f32_16x16x32_bf16(false, a, false, b,
                                                    (short)0, acc, false, false);
    }
    if (lane < 16) {
      int n = nt * 16 + lane;
#pragma unroll
      for (int r = 0; r < BB; ++r) sv[r][n] = acc[r];
    }
  }
  __syncthreads();

  // ---- stage 6: logits = m2 . relu(G@M1[512:] + v + c1) ----
  {
    float lp[4][8];
#pragma unroll
    for (int rt = 0; rt < 4; ++rt)
#pragma unroll
      for (int r = 0; r < 8; ++r) lp[rt][r] = 0.f;

    for (int ntb = 0; ntb < 16; ++ntb) {
      int nt = wv * 16 + ntb;
      v8f acc[4];
#pragma unroll
      for (int rt = 0; rt < 4; ++rt) acc[rt] = (v8f){};
      for (int kt = 0; kt < KT; ++kt) {
        v16bf b = *(const v16bf*)&M1b[((size_t)(nt * KT + kt)) * 512 + lane * 16];
#pragma unroll
        for (int rt = 0; rt < 4; ++rt) {
          v16bf a = *(const v16bf*)&sG[(rt * KT + kt) * 512 + lane * 16];
          acc[rt] = __builtin_amdgcn_wmma_f32_16x16x32_bf16(
              false, a, false, b, (short)0, acc[rt], false, false);
        }
      }
      int n = nt * 16 + (lane & 15);
      float m2n = sm2[n];
#pragma unroll
      for (int rt = 0; rt < 4; ++rt) {
        float cv = sv[rt][n] + sc1[n];
#pragma unroll
        for (int r = 0; r < 8; ++r) {
          float h = fmaxf(acc[rt][r] + cv, 0.f);
          lp[rt][r] += h * m2n;
        }
      }
    }
#pragma unroll
    for (int rt = 0; rt < 4; ++rt)
#pragma unroll
      for (int r = 0; r < 8; ++r)
        atomicAdd(&slog[rt][r + 8 * half], lp[rt][r]);
  }
  __syncthreads();

  // ---- stage 7: emit logits ----
  if (tid < BB * KSP) {
    int bb = tid >> 4, k = tid & 15;
    out[(size_t)(b0 + bb) * KSP + k] = slog[bb][k] + c2p[0];
  }
}

// ---------------------------------------------------------------------------
extern "C" void kernel_launch(void* const* d_in, const int* in_sizes, int n_in,
                              void* d_out, int out_size, void* d_ws, size_t ws_size,
                              hipStream_t stream) {
  const float* X  = (const float*)d_in[0];
  const float* G  = (const float*)d_in[1];
  const float* W1 = (const float*)d_in[2];
  const float* b1 = (const float*)d_in[3];
  const float* w2 = (const float*)d_in[4];
  // d_in[5] = b2 (softmax-shift-invariant, unused)
  const float* M1 = (const float*)d_in[6];
  const float* c1 = (const float*)d_in[7];
  const float* m2 = (const float*)d_in[8];
  const float* c2 = (const float*)d_in[9];
  float* out = (float*)d_out;

  // Workspace: packed bf16 weights, WMMA B-matrix layout. 6 MB total.
  unsigned short* P   = (unsigned short*)d_ws;
  unsigned short* W1a = P;                         // 64 nt * 16 kt * 512
  unsigned short* W1b = P + 524288;
  unsigned short* M1a = P + 1048576;               // 128 nt * 16 kt * 512
  unsigned short* M1b = P + 2097152;

  pack_w_kernel<<<524288 / 256, 256, 0, stream>>>(W1, W1a, 64, 0,   1024);
  pack_w_kernel<<<524288 / 256, 256, 0, stream>>>(W1, W1b, 64, 512, 1024);
  pack_w_kernel<<<1048576 / 256, 256, 0, stream>>>(M1, M1a, 128, 0,   2048);
  pack_w_kernel<<<1048576 / 256, 256, 0, stream>>>(M1, M1b, 128, 512, 2048);

  guesser_fused_kernel<<<4096 / BB, 256, 0, stream>>>(
      X, G, b1, w2, c1, m2, c2, W1a, W1b, M1a, M1b, out);
}
